// SmallObjectDetectionOptimizer_16423954940592
// MI455X (gfx1250) — compile-verified, matrix operand order checked by reading, not run on hardware
//
#include <hip/hip_runtime.h>
#include <math.h>

#define BATCH 4
#define CCH   256
#define BN_RSQRT 0.9999950000374998f   /* 1/sqrt(1+1e-5) */

typedef __attribute__((ext_vector_type(16))) __bf16 v16bf;
typedef __attribute__((ext_vector_type(8)))  float  v8f;

union FragU { v16bf v; uint4 q[2]; unsigned short u[16]; };

__device__ __forceinline__ unsigned short f2bf(float f) {
  union { float f; unsigned u; } x; x.f = f;
  unsigned r = x.u + 0x7FFFu + ((x.u >> 16) & 1u);   // round-to-nearest-even
  return (unsigned short)(r >> 16);
}
__device__ __forceinline__ float bf2f(unsigned short h) {
  union { unsigned u; float f; } x; x.u = ((unsigned)h) << 16; return x.f;
}
__device__ __forceinline__ float sigmoidf_(float x) { return 1.f / (1.f + __expf(-x)); }

// K offset of half-VGPR slot j for lane-group hi (ISA 7.12.2, 16-bit 16x32 frag)
__device__ __forceinline__ int kofs(int j, int hi) {
  return (j < 4) ? (8 * hi + 2 * j) : (16 + 8 * hi + 2 * (j - 4));
}

// ---------------------------------------------------------------------------
// Weight pre-pack: W[M][K] fp32 -> bf16 fragment order [M/16][K/32][32][16].
// ---------------------------------------------------------------------------
__global__ void pack_w_kernel(const float* __restrict__ W,
                              unsigned short* __restrict__ OUT, int M, int K) {
  const int t = blockIdx.x * blockDim.x + threadIdx.x;
  const int nk = K >> 5;
  if (t >= (M >> 4) * nk * 32) return;
  const int lane = t & 31, tile = t >> 5;
  const int kc = tile % nk, mt = tile / nk;
  const int lo = lane & 15, hi = lane >> 4;
  const float* wr = W + (size_t)(mt * 16 + lo) * K + kc * 32;
  unsigned short* o = OUT + (size_t)t * 16;
#pragma unroll
  for (int j = 0; j < 8; ++j) {
    const int k = kofs(j, hi);
    o[2 * j]     = f2bf(wr[k]);
    o[2 * j + 1] = f2bf(wr[k + 1]);
  }
}

// 3x3 weights W[256][256][3][3] -> [9][16][8][32][16]
__global__ void pack_w3_kernel(const float* __restrict__ W,
                               unsigned short* __restrict__ OUT) {
  const int t = blockIdx.x * blockDim.x + threadIdx.x;
  if (t >= 9 * 16 * 8 * 32) return;
  const int lane = t & 31;
  int r = t >> 5;
  const int kc = r & 7; r >>= 3;
  const int mt = r & 15;
  const int tap = r >> 4;
  const int lo = lane & 15, hi = lane >> 4;
  const float* wr = W + ((size_t)(mt * 16 + lo) * CCH + kc * 32) * 9 + tap;
  unsigned short* o = OUT + (size_t)t * 16;
#pragma unroll
  for (int j = 0; j < 8; ++j) {
    const int k = kofs(j, hi);
    o[2 * j]     = f2bf(wr[(size_t)k * 9]);
    o[2 * j + 1] = f2bf(wr[(size_t)(k + 1) * 9]);
  }
}

// NCHW fp32 -> NHWC bf16 via LDS-tiled transpose (coalesced both sides).
__global__ void nchw2nhwc_kernel(const float* __restrict__ IN,
                                 unsigned short* __restrict__ OUT, int P) {
  __shared__ float tile[32][33];
  const int b = blockIdx.z;
  const int c0 = blockIdx.y * 32, p0 = blockIdx.x * 32;
  const int tx = threadIdx.x & 31, ty = threadIdx.x >> 5;   // 32 x 8
#pragma unroll
  for (int i = 0; i < 4; ++i) {
    const int c = c0 + ty + i * 8, p = p0 + tx;
    tile[ty + i * 8][tx] = (p < P) ? IN[((size_t)b * CCH + c) * P + p] : 0.f;
  }
  __syncthreads();
#pragma unroll
  for (int i = 0; i < 4; ++i) {
    const int p = p0 + ty + i * 8, c = c0 + tx;
    if (p < P) OUT[((size_t)b * P + p) * CCH + c] = f2bf(tile[tx][ty + i * 8]);
  }
}

__global__ void zero_kernel(uint4* __restrict__ p, size_t n16) {
  const size_t i = (size_t)blockIdx.x * blockDim.x + threadIdx.x;
  if (i < n16) { uint4 z; z.x = z.y = z.z = z.w = 0u; p[i] = z; }
}

// ---------------------------------------------------------------------------
// WMMA GEMM, NHWC activations, pre-packed bf16 weights.
// Per K-chunk: one clause of (2 A + NT*2 B) b128 loads, then NT WMMAs.
// Loads return in order -> compiler can use partial loadcnt waits and overlap
// the clause tail with the first WMMAs.
// ---------------------------------------------------------------------------
template <int NT, bool BNRELU, bool PADOUT>
__global__ void gemm_wmma_kernel(const unsigned short* __restrict__ PA,
                                 const unsigned short* __restrict__ IN,   // [NPIX][K]
                                 unsigned short* __restrict__ OUT,
                                 const float* __restrict__ gamma,
                                 const float* __restrict__ beta,
                                 int M, int K, int NPIX, int P, int S, int Sp) {
  const int lane = threadIdx.x;
  const int lo = lane & 15, hi = lane >> 4;
  const int mt = blockIdx.y, oc0 = mt * 16;
  const int t0 = blockIdx.x * NT;
  const int ntiles = NPIX >> 4;   // NPIX is always a multiple of 16
  const int nk = K >> 5;
  const unsigned short* pa = PA + ((size_t)mt * nk * 32 + lane) * 16;
  const unsigned short* pb[NT];
  int gcol[NT];
#pragma unroll
  for (int s = 0; s < NT; ++s) {
    const int tt = min(t0 + s, ntiles - 1);   // clamp loads; stores guarded below
    gcol[s] = tt * 16 + lo;
    pb[s] = IN + (size_t)gcol[s] * K + 8 * hi;
  }
  v8f acc[NT] = {};
  for (int kk = 0; kk < nk; ++kk) {
    FragU a, bf[NT];
    a.q[0] = *(const uint4*)(pa);
    a.q[1] = *(const uint4*)(pa + 8);
    pa += 32 * 16;
#pragma unroll
    for (int s = 0; s < NT; ++s) {
      bf[s].q[0] = *(const uint4*)(pb[s]);
      bf[s].q[1] = *(const uint4*)(pb[s] + 16);
      pb[s] += 32;
    }
#pragma unroll
    for (int s = 0; s < NT; ++s)
      acc[s] = __builtin_amdgcn_wmma_f32_16x16x32_bf16(false, a.v, false, bf[s].v,
                                                       (short)0, acc[s], false, false);
  }
  float gg[8], bb[8];
  if (BNRELU) {
#pragma unroll
    for (int r = 0; r < 8; ++r) {
      gg[r] = gamma[oc0 + 8 * hi + r] * BN_RSQRT;
      bb[r] = beta[oc0 + 8 * hi + r];
    }
  }
#pragma unroll
  for (int s = 0; s < NT; ++s) {
    if (t0 + s >= ntiles) break;
    FragU o;
#pragma unroll
    for (int r = 0; r < 8; ++r) {       // D: M = r + 8*hi, N = lo
      float v = acc[s][r];
      if (BNRELU) v = fmaxf(v * gg[r] + bb[r], 0.f);
      o.u[r] = f2bf(v);
    }
    size_t idx;
    const int g = gcol[s];
    if (PADOUT) {
      const int b = g / P, p = g - b * P;
      const int y = p / S, x = p - y * S;
      idx = (((size_t)b * Sp + y + 1) * Sp + (x + 1)) * (size_t)M + oc0 + 8 * hi;
    } else {
      idx = (size_t)g * M + oc0 + 8 * hi;
    }
    *(uint4*)(OUT + idx) = o.q[0];
  }
}

// 3x3 conv (pad 1) over zero-haloed NHWC input: taps are uniform scalar offsets.
template <int NT>
__global__ void conv3_wmma_kernel(const unsigned short* __restrict__ PA, // [9][16][8][32][16]
                                  const unsigned short* __restrict__ IN, // [B][Sp][Sp][256]
                                  unsigned short* __restrict__ OUT,      // [NPIX][256]
                                  const float* __restrict__ gamma,
                                  const float* __restrict__ beta,
                                  int P, int S, int Sp) {
  const int lane = threadIdx.x;
  const int lo = lane & 15, hi = lane >> 4;
  const int mt = blockIdx.y, oc0 = mt * 16;
  const int t0 = blockIdx.x * NT;
  const int ntiles = (BATCH * P) >> 4;
  const unsigned short* pb[NT];
  int gcol[NT];
#pragma unroll
  for (int s = 0; s < NT; ++s) {
    const int tt = min(t0 + s, ntiles - 1);
    const int g = tt * 16 + lo;
    gcol[s] = g;
    const int b = g / P, p = g - b * P;
    const int y = p / S, x = p - y * S;
    pb[s] = IN + (((size_t)b * Sp + y + 1) * Sp + (x + 1)) * CCH + 8 * hi;
  }
  v8f acc[NT] = {};
  for (int t = 0; t < 9; ++t) {
    const int toff = ((t / 3 - 1) * Sp + (t % 3 - 1)) * CCH;   // uniform
    const unsigned short* pa = PA + (((size_t)t * 16 + mt) * 8 * 32 + lane) * 16;
    for (int kk = 0; kk < 8; ++kk) {
      FragU a, bf[NT];
      a.q[0] = *(const uint4*)(pa);
      a.q[1] = *(const uint4*)(pa + 8);
      pa += 512;
#pragma unroll
      for (int s = 0; s < NT; ++s) {
        const unsigned short* src = pb[s] + toff + kk * 32;
        bf[s].q[0] = *(const uint4*)(src);
        bf[s].q[1] = *(const uint4*)(src + 16);
      }
#pragma unroll
      for (int s = 0; s < NT; ++s)
        acc[s] = __builtin_amdgcn_wmma_f32_16x16x32_bf16(false, a.v, false, bf[s].v,
                                                         (short)0, acc[s], false, false);
    }
  }
  float gg[8], bb[8];
#pragma unroll
  for (int r = 0; r < 8; ++r) {
    gg[r] = gamma[oc0 + 8 * hi + r] * BN_RSQRT;
    bb[r] = beta[oc0 + 8 * hi + r];
  }
#pragma unroll
  for (int s = 0; s < NT; ++s) {
    if (t0 + s >= ntiles) break;
    FragU o;
#pragma unroll
    for (int r = 0; r < 8; ++r)
      o.u[r] = f2bf(fmaxf(acc[s][r] * gg[r] + bb[r], 0.f));
    *(uint4*)(OUT + (size_t)gcol[s] * CCH + oc0 + 8 * hi) = o.q[0];
  }
}

// Last enhancer GEMM (K=64) + sigmoid, fused out = xsa * sigmoid(.), fp32 NCHW.
template <int NT>
__global__ void final_wmma_kernel(const unsigned short* __restrict__ PA,  // [16][2][32][16]
                                  const unsigned short* __restrict__ MID, // [NPIX][64]
                                  const unsigned short* __restrict__ XSA, // [NPIX][256]
                                  float* __restrict__ OUT, int P) {
  const int lane = threadIdx.x;
  const int lo = lane & 15, hi = lane >> 4;
  const int mt = blockIdx.y, oc0 = mt * 16;
  const int t0 = blockIdx.x * NT;
  const int ntiles = (BATCH * P) >> 4;
  const unsigned short* pb[NT];
  int gcol[NT];
#pragma unroll
  for (int s = 0; s < NT; ++s) {
    const int tt = min(t0 + s, ntiles - 1);
    gcol[s] = tt * 16 + lo;
    pb[s] = MID + (size_t)gcol[s] * 64 + 8 * hi;
  }
  v8f acc[NT] = {};
  const unsigned short* pa = PA + ((size_t)mt * 2 * 32 + lane) * 16;
#pragma unroll
  for (int kk = 0; kk < 2; ++kk) {
    FragU a, bf[NT];
    a.q[0] = *(const uint4*)(pa);
    a.q[1] = *(const uint4*)(pa + 8);
    pa += 512;
#pragma unroll
    for (int s = 0; s < NT; ++s) {
      bf[s].q[0] = *(const uint4*)(pb[s]);
      bf[s].q[1] = *(const uint4*)(pb[s] + 16);
      pb[s] += 32;
    }
#pragma unroll
    for (int s = 0; s < NT; ++s)
      acc[s] = __builtin_amdgcn_wmma_f32_16x16x32_bf16(false, a.v, false, bf[s].v,
                                                       (short)0, acc[s], false, false);
  }
#pragma unroll
  for (int s = 0; s < NT; ++s) {
    if (t0 + s >= ntiles) break;
    const int g = gcol[s];
    const int b = g / P, p = g - b * P;
    FragU x;
    x.q[0] = *(const uint4*)(XSA + (size_t)g * CCH + oc0 + 8 * hi);
#pragma unroll
    for (int r = 0; r < 8; ++r) {
      const int oc = oc0 + 8 * hi + r;
      OUT[(size_t)(b * CCH + oc) * P + p] = bf2f(x.u[r]) * sigmoidf_(acc[s][r]);
    }
  }
}

// Bilinear x2 upsample + add, NHWC bf16 (consecutive threads = consecutive c).
__global__ void upsample_add_kernel(const unsigned short* __restrict__ SRC,
                                    unsigned short* __restrict__ DST,
                                    int sS, int sD) {
  const int Pd = sD * sD, Ps = sS * sS;
  const size_t e = (size_t)blockIdx.x * blockDim.x + threadIdx.x;
  if (e >= (size_t)BATCH * Pd * CCH) return;
  const int c = (int)(e & (CCH - 1));
  const int g = (int)(e >> 8);
  const int b = g / Pd, p = g - b * Pd;
  const int y = p / sD, x = p - y * sD;
  const float sy = (y + 0.5f) * 0.5f - 0.5f;
  const float sx = (x + 0.5f) * 0.5f - 0.5f;
  int y0 = (int)floorf(sy); const float ty = sy - y0;
  int x0 = (int)floorf(sx); const float tx = sx - x0;
  const int y1 = min(y0 + 1, sS - 1); y0 = max(y0, 0);
  const int x1 = min(x0 + 1, sS - 1); x0 = max(x0, 0);
  const unsigned short* sp = SRC + (size_t)b * Ps * CCH + c;
  const float v00 = bf2f(sp[(size_t)(y0 * sS + x0) * CCH]);
  const float v01 = bf2f(sp[(size_t)(y0 * sS + x1) * CCH]);
  const float v10 = bf2f(sp[(size_t)(y1 * sS + x0) * CCH]);
  const float v11 = bf2f(sp[(size_t)(y1 * sS + x1) * CCH]);
  const float up = (1.f - ty) * ((1.f - tx) * v00 + tx * v01) +
                   ty * ((1.f - tx) * v10 + tx * v11);
  DST[e] = f2bf(bf2f(DST[e]) + up);
}

// Partial GAP sums (NHWC, thread=channel -> coalesced), accumulate via atomics.
__global__ void gap_kernel(const unsigned short* __restrict__ O,
                           float* __restrict__ gap, int P, int nch) {
  const int b = blockIdx.x / nch;
  const int p0 = (blockIdx.x - b * nch) * 256;
  const int c = threadIdx.x;
  const int pend = min(p0 + 256, P);
  const unsigned short* base = O + (size_t)b * P * CCH + c;
  float s = 0.f;
  for (int p = p0; p < pend; ++p) s += bf2f(base[(size_t)p * CCH]);
  atomicAdd(&gap[b * CCH + c], s);
}

// Channel-attention MLP 256->16(ReLU)->256(sigmoid); gap holds sums (scale invP).
__global__ void ca_kernel(const float* __restrict__ gap,
                          const float* __restrict__ w1, const float* __restrict__ w2,
                          float* __restrict__ ca, float invP) {
  __shared__ float hid[16];
  const int t = threadIdx.x;   // 256
  for (int b = 0; b < BATCH; ++b) {
    if (t < 16) {
      float s = 0.f;
      for (int c = 0; c < CCH; ++c) s += w1[t * CCH + c] * gap[b * CCH + c];
      hid[t] = fmaxf(s * invP, 0.f);
    }
    __syncthreads();
    float s = 0.f;
#pragma unroll
    for (int h = 0; h < 16; ++h) s += w2[t * 16 + h] * hid[h];
    ca[b * CCH + t] = sigmoidf_(s);
    __syncthreads();
  }
}

// Per-pixel mean & max over channels of o*ca (NHWC: 256 contiguous bf16/pixel).
__global__ void meanmax_kernel(const unsigned short* __restrict__ O,
                               const float* __restrict__ ca,
                               float* __restrict__ avg, float* __restrict__ mx, int P) {
  const int g = blockIdx.x * blockDim.x + threadIdx.x;
  if (g >= BATCH * P) return;
  const int b = g / P;
  const unsigned short* src = O + (size_t)g * CCH;
  const float* cab = ca + b * CCH;
  float s = 0.f, m = -3.4e38f;
  for (int c = 0; c < CCH; c += 8) {
    uint4 q = *(const uint4*)(src + c);
    const unsigned short* h = (const unsigned short*)&q;
#pragma unroll
    for (int j = 0; j < 8; ++j) {
      const float v = bf2f(h[j]) * cab[c + j];
      s += v; m = fmaxf(m, v);
    }
  }
  avg[g] = s * (1.f / CCH);
  mx[g] = m;
}

// 7x7 spatial-attention conv (pad 3, zero) + sigmoid.
__global__ void sa_kernel(const float* __restrict__ avg, const float* __restrict__ mx,
                          const float* __restrict__ w, float* __restrict__ sa,
                          int S, int P) {
  const int g = blockIdx.x * blockDim.x + threadIdx.x;
  if (g >= BATCH * P) return;
  const int b = g / P, p = g - b * P;
  const int y = p / S, x = p - y * S;
  const float* a = avg + b * P;
  const float* m2 = mx + b * P;
  float acc = 0.f;
  for (int ky = 0; ky < 7; ++ky) {
    const int yy = y + ky - 3;
    if (yy < 0 || yy >= S) continue;
    for (int kx = 0; kx < 7; ++kx) {
      const int xx = x + kx - 3;
      if (xx < 0 || xx >= S) continue;
      const int q = yy * S + xx;
      acc += w[ky * 7 + kx] * a[q] + w[49 + ky * 7 + kx] * m2[q];
    }
  }
  sa[g] = sigmoidf_(acc);
}

// xsa = o * ca[c] * sa[p]  (NHWC bf16), consumed by both enhancer GEMMs.
__global__ void xsa_kernel(const unsigned short* __restrict__ O,
                           const float* __restrict__ ca, const float* __restrict__ sa,
                           unsigned short* __restrict__ X, int P) {
  const size_t e = (size_t)blockIdx.x * blockDim.x + threadIdx.x;
  if (e >= (size_t)BATCH * P * CCH) return;
  const int c = (int)(e & (CCH - 1));
  const int g = (int)(e >> 8);
  const int b = g / P;
  X[e] = f2bf(bf2f(O[e]) * ca[b * CCH + c] * sa[g]);
}

// ---------------------------------------------------------------------------
extern "C" void kernel_launch(void* const* d_in, const int* in_sizes, int n_in,
                              void* d_out, int out_size, void* d_ws, size_t ws_size,
                              hipStream_t stream) {
  static const int Sdim[5] = {160, 80, 40, 20, 10};
  static const int Ps[5]   = {25600, 6400, 1600, 400, 100};
  static const size_t cum[6]    = {0, 25600, 32000, 33600, 34000, 34100};
  static const size_t padcum[6] = {0, 26244, 32968, 34732, 35216, 35360}; // (S+2)^2 cums

  const float* feat[5];
  for (int i = 0; i < 5; ++i) feat[i] = (const float*)d_in[i];
  const float* lat_w = (const float*)d_in[5];
  const float* enh_w = (const float*)d_in[6];
  const float* enh_g = (const float*)d_in[7];
  const float* enh_b = (const float*)d_in[8];
  const float* out_w = (const float*)d_in[9];
  const float* out_g = (const float*)d_in[10];
  const float* out_b = (const float*)d_in[11];
  const float* ca_w1 = (const float*)d_in[12];
  const float* ca_w2 = (const float*)d_in[13];
  const float* sa_w  = (const float*)d_in[14];
  const float* en_w1 = (const float*)d_in[15];
  const float* en_g  = (const float*)d_in[16];
  const float* en_b  = (const float*)d_in[17];
  const float* en_w2 = (const float*)d_in[18];

  char* ws = (char*)d_ws;
  const size_t ETOT = (size_t)BATCH * CCH * 34100;
  unsigned short* featT = (unsigned short*)ws; ws += ETOT * 2;  // aliased as xsa later
  unsigned short* lat   = (unsigned short*)ws; ws += ETOT * 2;
  unsigned short* epad  = (unsigned short*)ws; ws += (size_t)BATCH * CCH * 35360 * 2;
  unsigned short* obuf  = (unsigned short*)ws; ws += ETOT * 2;
  unsigned short* mid   = (unsigned short*)ws; ws += (size_t)BATCH * 64 * 34100 * 2;
  unsigned short* p_lat = (unsigned short*)ws; ws += 5 * 65536 * 2;
  unsigned short* p_enh = (unsigned short*)ws; ws += 5 * 65536 * 2;
  unsigned short* p_out = (unsigned short*)ws; ws += (size_t)5 * 589824 * 2;
  unsigned short* p_en1 = (unsigned short*)ws; ws += 16384 * 2;
  unsigned short* p_en2 = (unsigned short*)ws; ws += 16384 * 2;
  float* gapb = (float*)ws; ws += 1024 * 4;
  float* cab  = (float*)ws; ws += 1024 * 4;
  float* avgb = (float*)ws; ws += (size_t)BATCH * 34100 * 4;
  float* mxb  = (float*)ws; ws += (size_t)BATCH * 34100 * 4;
  float* sab  = (float*)ws; ws += (size_t)BATCH * 34100 * 4;
  unsigned short* xsa = featT;   // feats NHWC no longer needed after laterals

  const int NT = 4;

  // 0) pre-pack weights (bf16 fragment order) + transpose inputs to NHWC bf16
  for (int i = 0; i < 5; ++i) {
    pack_w_kernel<<<16, 256, 0, stream>>>(lat_w + (size_t)i * 65536, p_lat + (size_t)i * 65536, CCH, CCH);
    pack_w_kernel<<<16, 256, 0, stream>>>(enh_w + (size_t)i * 65536, p_enh + (size_t)i * 65536, CCH, CCH);
    pack_w3_kernel<<<144, 256, 0, stream>>>(out_w + (size_t)i * 589824, p_out + (size_t)i * 589824);
    dim3 tg((Ps[i] + 31) / 32, 8, BATCH);
    nchw2nhwc_kernel<<<tg, 256, 0, stream>>>(feat[i], featT + 1024 * cum[i], Ps[i]);
  }
  pack_w_kernel<<<4, 256, 0, stream>>>(en_w1, p_en1, 64, CCH);
  pack_w_kernel<<<4, 256, 0, stream>>>(en_w2, p_en2, CCH, 64);

  // 1) lateral 1x1 convs
  for (int i = 0; i < 5; ++i) {
    const int NPIX = BATCH * Ps[i];
    dim3 g((NPIX / 16 + NT - 1) / NT, 16);
    gemm_wmma_kernel<NT, false, false><<<g, 32, 0, stream>>>(
        p_lat + (size_t)i * 65536, featT + 1024 * cum[i], lat + 1024 * cum[i],
        nullptr, nullptr, CCH, CCH, NPIX, Ps[i], Sdim[i], 0);
  }
  // 2) top-down bilinear fusion
  for (int i = 3; i >= 0; --i) {
    const size_t n = (size_t)BATCH * Ps[i] * CCH;
    upsample_add_kernel<<<(unsigned)((n + 255) / 256), 256, 0, stream>>>(
        lat + 1024 * cum[i + 1], lat + 1024 * cum[i], Sdim[i + 1], Sdim[i]);
  }
  // 3-5) per-level heads + attention
  for (int i = 0; i < 5; ++i) {
    const int P = Ps[i], S = Sdim[i], Sp = S + 2;
    const int NPIX = BATCH * P;
    unsigned short* epadi = epad + 1024 * padcum[i];
    unsigned short* oi    = obuf + 1024 * cum[i];
    unsigned short* midi  = mid + 256 * cum[i];
    unsigned short* xsai  = xsa + 1024 * cum[i];
    float* outi = (float*)d_out + 1024 * cum[i];
    dim3 g((NPIX / 16 + NT - 1) / NT, 16);
    dim3 g64((NPIX / 16 + NT - 1) / NT, 4);

    const size_t padn16 = (size_t)BATCH * Sp * Sp * CCH * 2 / 16;
    zero_kernel<<<(unsigned)((padn16 + 255) / 256), 256, 0, stream>>>((uint4*)epadi, padn16);

    gemm_wmma_kernel<NT, true, true><<<g, 32, 0, stream>>>(
        p_enh + (size_t)i * 65536, lat + 1024 * cum[i], epadi,
        enh_g + i * CCH, enh_b + i * CCH, CCH, CCH, NPIX, P, S, Sp);

    conv3_wmma_kernel<NT><<<g, 32, 0, stream>>>(
        p_out + (size_t)i * 589824, epadi, oi, out_g + i * CCH, out_b + i * CCH, P, S, Sp);

    zero_kernel<<<1, 256, 0, stream>>>((uint4*)gapb, 256);
    const int nch = (P + 255) / 256;
    gap_kernel<<<BATCH * nch, 256, 0, stream>>>(oi, gapb, P, nch);
    ca_kernel<<<1, 256, 0, stream>>>(gapb, ca_w1, ca_w2, cab, 1.f / (float)P);
    meanmax_kernel<<<(NPIX + 255) / 256, 256, 0, stream>>>(oi, cab, avgb, mxb, P);
    sa_kernel<<<(NPIX + 255) / 256, 256, 0, stream>>>(avgb, mxb, sa_w, sab, S, P);
    const size_t ne = (size_t)NPIX * CCH;
    xsa_kernel<<<(unsigned)((ne + 255) / 256), 256, 0, stream>>>(oi, cab, sab, xsai, P);

    gemm_wmma_kernel<NT, true, false><<<g64, 32, 0, stream>>>(
        p_en1, xsai, midi, en_g, en_b, 64, CCH, NPIX, P, S, 0);

    final_wmma_kernel<NT><<<g, 32, 0, stream>>>(p_en2, midi, xsai, outi, P);
  }
}
